// ESNDriver_84902913507495
// MI455X (gfx1250) — matrix operand order repaired and checked by reading, not compiled, not run on hardware
//
#include <hip/hip_runtime.h>
#include <math.h>

#define RES_DIM 8192
#define BATCH   256
#define LEAK    0.6f
#define BIAS    1.6f

typedef __attribute__((ext_vector_type(16))) _Float16 v16h;
typedef __attribute__((ext_vector_type(8)))  float    v8f;

union Frag16 {          // 32 bytes = one 16-bit WMMA A/B fragment per lane
  uint4 u4[2];
  v16h  v;
};

// ---------------- kernel: zero CSR metadata ----------------
__global__ void k_zero(int* __restrict__ p, int n) {
  int i = blockIdx.x * blockDim.x + threadIdx.x;
  if (i < n) p[i] = 0;
}

// ---------------- kernel: per-row nnz histogram ----------------
__global__ void k_hist(const int* __restrict__ rows, int* __restrict__ counts, int nnz) {
  int e = blockIdx.x * blockDim.x + threadIdx.x;
  if (e < nnz) atomicAdd(&counts[rows[e]], 1);
}

// ---------------- kernel: exclusive scan (8192 counts -> row_ptr[8193]) ----------------
__global__ __launch_bounds__(256) void k_scan(const int* __restrict__ counts,
                                              int* __restrict__ row_ptr) {
  __shared__ int part[256];
  const int t = threadIdx.x;
  const int base = t * 32;
  int s = 0;
#pragma unroll
  for (int i = 0; i < 32; ++i) s += counts[base + i];
  part[t] = s;
  __syncthreads();
  if (t == 0) {
    int acc = 0;
    for (int i = 0; i < 256; ++i) { int v = part[i]; part[i] = acc; acc += v; }
    row_ptr[RES_DIM] = acc;
  }
  __syncthreads();
  int acc = part[t];
  for (int i = 0; i < 32; ++i) { row_ptr[base + i] = acc; acc += counts[base + i]; }
}

// ---------------- kernel: scatter COO -> CSR ----------------
__global__ void k_scatter(const int* __restrict__ rows, const int* __restrict__ cols,
                          const float* __restrict__ vals,
                          const int* __restrict__ row_ptr, int* __restrict__ fill,
                          int* __restrict__ ccols, float* __restrict__ cvals, int nnz) {
  int e = blockIdx.x * blockDim.x + threadIdx.x;
  if (e < nnz) {
    int r = rows[e];
    int pos = row_ptr[r] + atomicAdd(&fill[r], 1);
    ccols[pos] = cols[e];
    cvals[pos] = vals[e];
  }
}

// ---------------- kernel: transpose+convert res_state (BxR f32) -> sT (RxB f16) ----------------
__global__ __launch_bounds__(256) void k_transpose(const float* __restrict__ rs,
                                                   _Float16* __restrict__ sT) {
  __shared__ float tile[32][33];
  const int j0 = blockIdx.x * 32, b0 = blockIdx.y * 32;
  const int tx = threadIdx.x, ty = threadIdx.y;  // 32 x 8
#pragma unroll
  for (int yy = 0; yy < 4; ++yy)
    tile[ty + yy * 8][tx] = rs[(size_t)(b0 + ty + yy * 8) * RES_DIM + j0 + tx];
  __syncthreads();
#pragma unroll
  for (int yy = 0; yy < 4; ++yy)
    sT[(size_t)(j0 + ty + yy * 8) * BATCH + b0 + tx] = (_Float16)tile[tx][ty + yy * 8];
}

// ---------------- kernel: CSR SpMM via WMMA + fused tanh/leak epilogue ----------------
// One workgroup (8 waves) per output row i. Gathered sT rows are staged into an
// n-major (transposed) LDS tile so every WMMA fragment is a pair of ds_load_b128
// (B: lane holds consecutive K for its N; A: consecutive-K runs). Tile row stride
// = 40 halves (80 B): 16 B aligned, bank-conflict-free fragment loads.
// Column indices are stored pre-shifted as byte offsets so every staging
// global_load_b64 address is a single 32-bit v_add against an SGPR base (GVS).
#define TROW 40                       // halves per tile row (32 data + 8 pad)
__global__ __launch_bounds__(256) void k_spmm_wmma(
    const int* __restrict__ row_ptr, const int* __restrict__ ccols,
    const float* __restrict__ cvals, const _Float16* __restrict__ sT,
    const float* __restrict__ proj, const float* __restrict__ res_state,
    float* __restrict__ out) {
  __shared__ uint4 tileT4[BATCH * (TROW / 8)];          // 20 KB, n-major: [n][k]
  __shared__ __align__(16) _Float16 sV16[512];          // chunk values (fp16)
  __shared__ unsigned sC[512];                          // column byte offsets (j << 9)

  const int i    = blockIdx.x;
  const int t    = threadIdx.x;
  const int lane = t & 31;
  const int wave = t >> 5;          // 0..7
  const int n0a  = wave * 32;
  const int n0b  = wave * 32 + 16;

  const int start = row_ptr[i];
  const int end   = row_ptr[i + 1];

  // staging role: thread covers k-octet kg (0..3) x batch-quad nq (0..63)
  const int      kg    = t >> 6;
  const int      nq    = t & 63;
  const unsigned nqoff = (unsigned)nq * 8u;     // byte offset of this thread's 4 cols
  const char*    sTc   = (const char*)sT;       // uniform 64-bit base (SGPR pair)

  // fragment constants
  const int  n     = lane & 15;
  const int  kqo   = (lane < 16) ? 0 : 2;   // B: uint4 offset of K base {0,16}
  const int  koffA = (lane < 16) ? 0 : 16;  // A: K base (contiguous k order, same as B)
  const bool am0   = ((lane & 15) == 0);    // only M=0 row of A carries values

  v8f c0 = {}; v8f c1 = {};
  const v16h vzero = {};

  for (int s0 = start; s0 < end; s0 += 512) {
    // ---- stage up to 512 (val, col-byte-offset) pairs, zero-padded ----
#pragma unroll
    for (int rep = 0; rep < 2; ++rep) {
      int idx = t + rep * 256;
      int e = s0 + idx;
      float v = 0.f; unsigned cj = 0;
      if (e < end) { v = cvals[e]; cj = (unsigned)ccols[e] << 9; }
      sV16[idx] = (_Float16)v;
      sC[idx]   = cj;
    }
    __syncthreads();

    const int nloc = min(end - s0, 512);
    for (int e0 = 0; e0 < nloc; e0 += 32) {
      // ---- gather 32 sT rows into transposed LDS tile (saddr + 32-bit voffset) ----
      unsigned long long L[8];
#pragma unroll
      for (int k = 0; k < 8; ++k) {
        unsigned off = sC[e0 + kg * 8 + k] + nqoff;
        L[k] = *(const unsigned long long*)(sTc + off);
      }
#pragma unroll
      for (int q = 0; q < 4; ++q) {  // 8x4 register transpose, b128 store
        uint4 w;
        w.x = (unsigned)((L[0] >> (16 * q)) & 0xffffu) |
              ((unsigned)((L[1] >> (16 * q)) & 0xffffu) << 16);
        w.y = (unsigned)((L[2] >> (16 * q)) & 0xffffu) |
              ((unsigned)((L[3] >> (16 * q)) & 0xffffu) << 16);
        w.z = (unsigned)((L[4] >> (16 * q)) & 0xffffu) |
              ((unsigned)((L[5] >> (16 * q)) & 0xffffu) << 16);
        w.w = (unsigned)((L[6] >> (16 * q)) & 0xffffu) |
              ((unsigned)((L[7] >> (16 * q)) & 0xffffu) << 16);
        tileT4[(nq * 4 + q) * (TROW / 8) + kg] = w;
      }
      __syncthreads();

      // ---- A fragment: lane m holds K{0..15}, lane m+16 K{16..31} ----
      Frag16 fa; fa.v = vzero;
      if (am0) {
        const uint4* ap = (const uint4*)(sV16 + e0 + koffA);
        fa.u4[0] = ap[0];     // halves koffA .. +7
        fa.u4[1] = ap[1];     // halves koffA+8 .. +15
      }
      // ---- B fragments: lanes 0-15 hold K=0..15, lanes 16-31 K=16..31; N = lane%16 ----
      const uint4* tpa = tileT4 + (size_t)(n0a + n) * (TROW / 8);
      const uint4* tpb = tileT4 + (size_t)(n0b + n) * (TROW / 8);
      Frag16 fb0, fb1;
      fb0.u4[0] = tpa[kqo];  fb0.u4[1] = tpa[kqo + 1];
      fb1.u4[0] = tpb[kqo];  fb1.u4[1] = tpb[kqo + 1];

      c0 = __builtin_amdgcn_wmma_f32_16x16x32_f16(false, fa.v, false, fb0.v, (short)0, c0,
                                                  false, false);
      c1 = __builtin_amdgcn_wmma_f32_16x16x32_f16(false, fa.v, false, fb1.v, (short)0, c1,
                                                  false, false);
      __syncthreads();  // tile/metadata reuse
    }
  }

  // D row M=0 lives in VGPR0 (c[0]) on lanes 0..15, N = lane.
  if (lane < 16) {
    {
      int b = n0a + lane;
      size_t off = (size_t)b * RES_DIM + i;
      float rn = tanhf(c0[0] + proj[off] + BIAS);
      out[off] = LEAK * rn + (1.f - LEAK) * res_state[off];
    }
    {
      int b = n0b + lane;
      size_t off = (size_t)b * RES_DIM + i;
      float rn = tanhf(c1[0] + proj[off] + BIAS);
      out[off] = LEAK * rn + (1.f - LEAK) * res_state[off];
    }
  }
}

extern "C" void kernel_launch(void* const* d_in, const int* in_sizes, int n_in,
                              void* d_out, int out_size, void* d_ws, size_t ws_size,
                              hipStream_t stream) {
  const float* proj = (const float*)d_in[0];
  const float* rs   = (const float*)d_in[1];
  const int*   rows = (const int*)d_in[2];
  const int*   cols = (const int*)d_in[3];
  const float* vals = (const float*)d_in[4];
  float* out = (float*)d_out;
  const int nnz = in_sizes[2];
  (void)n_in; (void)out_size; (void)ws_size;

  char* ws = (char*)d_ws;
  size_t off = 0;
  _Float16* sT = (_Float16*)(ws + off); off += (size_t)RES_DIM * BATCH * 2;  // 4 MB
  int* counts  = (int*)(ws + off);      off += (size_t)RES_DIM * 4;
  int* fill    = (int*)(ws + off);      off += (size_t)RES_DIM * 4;          // contiguous w/ counts
  int* row_ptr = (int*)(ws + off);      off += (size_t)(RES_DIM + 8) * 4;
  int* ccols   = (int*)(ws + off);      off += (size_t)nnz * 4;
  float* cvals = (float*)(ws + off);    off += (size_t)nnz * 4;

  k_zero<<<(2 * RES_DIM + 255) / 256, 256, 0, stream>>>(counts, 2 * RES_DIM);
  k_hist<<<(nnz + 255) / 256, 256, 0, stream>>>(rows, counts, nnz);
  k_scan<<<1, 256, 0, stream>>>(counts, row_ptr);
  k_scatter<<<(nnz + 255) / 256, 256, 0, stream>>>(rows, cols, vals, row_ptr, fill,
                                                   ccols, cvals, nnz);
  k_transpose<<<dim3(RES_DIM / 32, BATCH / 32), dim3(32, 8), 0, stream>>>(rs, sT);
  k_spmm_wmma<<<RES_DIM, 256, 0, stream>>>(row_ptr, ccols, cvals, sT, proj, rs, out);
}
